// MoRBlock_34394098106471
// MI455X (gfx1250) — matrix operand order repaired and checked by reading, not compiled, not run on hardware
//
#include <hip/hip_runtime.h>
#include <hip/hip_bf16.h>
#include <stdint.h>

// ---------------------------------------------------------------------------
// MoR block for MI455X (gfx1250, wave32, WMMA).
// Compute-bound (~550 GFLOP selected-expert FFN vs ~30us of HBM traffic at
// 23.3 TB/s) -> bulk GEMMs via v_wmma_f32_16x16x32_bf16 (fp32 accumulate).
// Weights pre-packed N-major so both operands stage with b128 ops; tiles are
// double-buffered with one barrier per k-step; staging uses the CDNA5 async
// Global->LDS path (global_load_async_to_lds_b128 + s_wait_asynccnt) when the
// toolchain exposes it, else a VGPR round-trip fallback.
// ---------------------------------------------------------------------------

typedef __bf16 bf16_t;
typedef __attribute__((ext_vector_type(16))) __bf16 v16bf;
typedef __attribute__((ext_vector_type(8)))  float  v8f;

constexpr int Bb = 4, Ss = 4096, Hh = 1024, Ff = 4096, Ee = 8, Kk = 2;
constexpr int T  = Bb * Ss;     // 16384 tokens
constexpr int TK = T * Kk;      // 32768 (token, slot) pairs
constexpr float LN_EPS = 1e-5f;

union Frag16 { v16bf v; uint4 u[2]; };

// --- CDNA5 async Global->LDS staging (guarded; fallback keeps compile green)
#if defined(__has_builtin)
# if __has_builtin(__builtin_amdgcn_global_load_async_to_lds_b128)
#  define MOR_ASYNC_LDS 1
# endif
#endif
#ifndef MOR_ASYNC_LDS
# define MOR_ASYNC_LDS 0
#endif

#if MOR_ASYNC_LDS
typedef __attribute__((__vector_size__(4 * sizeof(int)))) int mor_v4i;
typedef __attribute__((address_space(1))) mor_v4i* mor_gv4i_p;   // global
typedef __attribute__((address_space(3))) mor_v4i* mor_lv4i_p;   // LDS
#endif

__device__ __forceinline__ void async_b128(const void* g, void* l) {
#if MOR_ASYNC_LDS
    __builtin_amdgcn_global_load_async_to_lds_b128(
        (mor_gv4i_p)(uintptr_t)g,
        (mor_lv4i_p)(unsigned)(uintptr_t)l,
        0, 0);
#else
    (void)g; (void)l;
#endif
}
__device__ __forceinline__ void wait_async_lds() {
#if MOR_ASYNC_LDS
# if defined(__has_builtin) && __has_builtin(__builtin_amdgcn_s_wait_asynccnt)
    __builtin_amdgcn_s_wait_asynccnt(0);
# else
    asm volatile("s_wait_asynccnt 0x0" ::: "memory");
# endif
#endif
}

// ---------------------------------------------------------------------------
// 0) zero per-expert counters
// ---------------------------------------------------------------------------
__global__ void init_counts_kernel(int* counts) {
    if (threadIdx.x < Ee) counts[threadIdx.x] = 0;
}

// ---------------------------------------------------------------------------
// 1) one-time weight pack: fp32 [E][R][C] -> bf16 [E][C][R] (N-major for the
//    GEMM B operand).  Tiled LDS transpose, both global streams coalesced.
// ---------------------------------------------------------------------------
__global__ __launch_bounds__(256) void transpose_pack_kernel(
    const float* __restrict__ src, bf16_t* __restrict__ dst, int R, int C) {
    __shared__ bf16_t tile[32][33];
    const int e = blockIdx.z;
    const float* s = src + (size_t)e * R * C;
    bf16_t*     d = dst + (size_t)e * R * C;
    const int c0 = blockIdx.x * 32, r0 = blockIdx.y * 32;
    const int tx = threadIdx.x & 31, ty = threadIdx.x >> 5;   // 32 x 8
#pragma unroll
    for (int i = 0; i < 32; i += 8)
        tile[ty + i][tx] = (bf16_t)s[(size_t)(r0 + ty + i) * C + c0 + tx];
    __syncthreads();
#pragma unroll
    for (int i = 0; i < 32; i += 8)
        d[(size_t)(c0 + ty + i) * R + r0 + tx] = tile[tx][ty + i];
}

// ---------------------------------------------------------------------------
// 2) fused LayerNorm + router + softmax + top-2.  One 256-thread block/token.
// ---------------------------------------------------------------------------
__global__ __launch_bounds__(256) void ln_router_kernel(
    const float* __restrict__ x, const float* __restrict__ gmm,
    const float* __restrict__ bta, const float* __restrict__ Wr,
    const float* __restrict__ br,
    bf16_t* __restrict__ xnb, float* __restrict__ scores,
    int* __restrict__ topi, int* __restrict__ counts) {
    const int t = blockIdx.x, tid = threadIdx.x;
    const float* xr = x + (size_t)t * Hh;
    __shared__ float red[256];
    __shared__ float slog[Ee];

    float v4[4];
    float s = 0.f, ss = 0.f;
#pragma unroll
    for (int i = 0; i < 4; ++i) {
        float v = xr[tid + 256 * i];
        v4[i] = v; s += v; ss += v * v;
    }
    red[tid] = s; __syncthreads();
    for (int o = 128; o > 0; o >>= 1) { if (tid < o) red[tid] += red[tid + o]; __syncthreads(); }
    const float mean = red[0] * (1.0f / Hh);
    __syncthreads();
    red[tid] = ss; __syncthreads();
    for (int o = 128; o > 0; o >>= 1) { if (tid < o) red[tid] += red[tid + o]; __syncthreads(); }
    const float var = red[0] * (1.0f / Hh) - mean * mean;   // biased, like jnp.var
    const float rstd = rsqrtf(var + LN_EPS);
    __syncthreads();

    float lg[Ee];
#pragma unroll
    for (int e = 0; e < Ee; ++e) lg[e] = 0.f;
#pragma unroll
    for (int i = 0; i < 4; ++i) {
        const int h = tid + 256 * i;
        const float xn = (v4[i] - mean) * rstd * gmm[h] + bta[h];
        xnb[(size_t)t * Hh + h] = (bf16_t)xn;
#pragma unroll
        for (int e = 0; e < Ee; ++e) lg[e] += xn * Wr[h * Ee + e];
    }
    for (int e = 0; e < Ee; ++e) {
        red[tid] = lg[e]; __syncthreads();
        for (int o = 128; o > 0; o >>= 1) { if (tid < o) red[tid] += red[tid + o]; __syncthreads(); }
        if (tid == 0) slog[e] = red[0] + br[e];
        __syncthreads();
    }
    if (tid == 0) {
        float mx = slog[0];
        for (int e = 1; e < Ee; ++e) mx = fmaxf(mx, slog[e]);
        float p[Ee]; float den = 0.f;
        for (int e = 0; e < Ee; ++e) { p[e] = __expf(slog[e] - mx); den += p[e]; }
        const float inv = 1.f / den;
        for (int e = 0; e < Ee; ++e) p[e] *= inv;
        int i0 = 0;
        for (int e = 1; e < Ee; ++e) if (p[e] > p[i0]) i0 = e;
        int i1 = (i0 == 0) ? 1 : 0;
        for (int e = 0; e < Ee; ++e) if (e != i0 && p[e] > p[i1]) i1 = e;
        scores[2 * t] = p[i0]; scores[2 * t + 1] = p[i1];
        topi[2 * t] = i0;      topi[2 * t + 1] = i1;
        atomicAdd(&counts[i0], 1);
        atomicAdd(&counts[i1], 1);
    }
}

// ---------------------------------------------------------------------------
// 3) exclusive prefix over 8 counts, zero the second counter bank
// ---------------------------------------------------------------------------
__global__ void offsets_kernel(const int* counts, int* offs, int* cnt2) {
    if (threadIdx.x == 0) {
        int o = 0;
        for (int e = 0; e < Ee; ++e) { offs[e] = o; o += counts[e]; cnt2[e] = 0; }
    }
}

// ---------------------------------------------------------------------------
// 4) build per-expert token lists; pos[t][k] -> row in act/Y
// ---------------------------------------------------------------------------
__global__ void build_list_kernel(const int* __restrict__ topi,
                                  const int* __restrict__ offs,
                                  int* cnt2, int* __restrict__ tok,
                                  int* __restrict__ pos) {
    const int t = blockIdx.x * blockDim.x + threadIdx.x;
    if (t >= T) return;
#pragma unroll
    for (int k = 0; k < Kk; ++k) {
        const int e = topi[2 * t + k];
        const int p = offs[e] + atomicAdd(&cnt2[e], 1);
        tok[p] = t;
        pos[2 * t + k] = p;
    }
}

// ---------------------------------------------------------------------------
// 5) c[e][h] = relu(b1[e]) @ W2[e] + b2[e]  (fp32, tiny)
// ---------------------------------------------------------------------------
__global__ void c_kernel(const float* __restrict__ b1, const float* __restrict__ W2,
                         const float* __restrict__ b2, float* __restrict__ c) {
    const int idx = blockIdx.x * blockDim.x + threadIdx.x;
    if (idx >= Ee * Hh) return;
    const int e = idx / Hh, h = idx % Hh;
    float acc = b2[idx];
    const float* w  = W2 + (size_t)e * Ff * Hh + h;
    const float* bb = b1 + (size_t)e * Ff;
    for (int f = 0; f < Ff; ++f) {
        float a = bb[f]; a = a > 0.f ? a : 0.f;
        acc += a * w[(size_t)f * Hh];
    }
    c[idx] = acc;
}

__global__ void ctot_kernel(const float* __restrict__ c, float* __restrict__ ct) {
    const int h = blockIdx.x * blockDim.x + threadIdx.x;
    if (h >= Hh) return;
    float s = 0.f;
    for (int e = 0; e < Ee; ++e) s += c[e * Hh + h];
    ct[h] = s;
}

// ---------------------------------------------------------------------------
// 6) GEMM1: act[p, f] = relu( xn[tok[p]] @ W1[e] + b1[e] )   (bf16 out)
//    128x128 tile, Kstep=32, double-buffered LDS, 8 waves -> each wave owns
//    64x32 of C as a 4x2 grid of v_wmma_f32_16x16x32_bf16 accumulators.
//    All staging base addresses are hoisted out of the k-loop.
// ---------------------------------------------------------------------------
__global__ __launch_bounds__(256) void gemm1_wmma_kernel(
    const bf16_t* __restrict__ xnb, const bf16_t* __restrict__ W1t,
    const float* __restrict__ b1, const int* __restrict__ tok,
    const int* __restrict__ counts, const int* __restrict__ offs,
    bf16_t* __restrict__ act) {
    const int e    = blockIdx.x >> 7;       // T/128 = 128 row-tiles per expert
    const int mt   = blockIdx.x & 127;
    const int n0   = blockIdx.y * 128;
    const int cnt  = counts[e];
    const int row0 = mt * 128;
    if (row0 >= cnt) return;
    const int mrows = min(128, cnt - row0);
    const int pbase = offs[e] + row0;

    __shared__ bf16_t As[2][128 * 32];      // A tile [m][k]
    __shared__ bf16_t Bs[2][128 * 32];      // B tile [n][k]  (pre-packed src)
    __shared__ int    stok[128];

    const int tid = threadIdx.x;
    if (tid < 128) stok[tid] = tok[pbase + ((tid < mrows) ? tid : 0)];
    __syncthreads();

    const int wid = tid >> 5, lane = tid & 31;
    const int wm = (wid & 1) * 64, wn = (wid >> 1) * 32;
    const int lrow = lane & 15, half = lane >> 4;

    v8f acc[4][2];
#pragma unroll
    for (int i = 0; i < 4; ++i)
#pragma unroll
        for (int j = 0; j < 2; ++j)
#pragma unroll
            for (int r = 0; r < 8; ++r) acc[i][j][r] = 0.0f;

    // hoisted staging bases (k-loop just adds kt*32 elements)
    const int sr = tid >> 1, sh = tid & 1;          // 128 rows x 2 half-rows(16)
    const bf16_t* aptr = xnb + (size_t)stok[sr] * Hh + sh * 16;
    const bf16_t* bptr = W1t + (size_t)e * Ff * Hh + (size_t)(n0 + sr) * Hh + sh * 16;
    bf16_t* adst[2] = { &As[0][sr * 32 + sh * 16], &As[1][sr * 32 + sh * 16] };
    bf16_t* bdst[2] = { &Bs[0][sr * 32 + sh * 16], &Bs[1][sr * 32 + sh * 16] };

    auto stage = [&](int kt, int buf) {
        const bf16_t* ak = aptr + kt * 32;
        const bf16_t* bk = bptr + kt * 32;
        __builtin_prefetch(bk + 64, 0, 0);          // weights 2 tiles ahead
#if MOR_ASYNC_LDS
        async_b128(ak,     adst[buf]);
        async_b128(ak + 8, adst[buf] + 8);
        async_b128(bk,     bdst[buf]);
        async_b128(bk + 8, bdst[buf] + 8);
#else
        const uint4* a4 = reinterpret_cast<const uint4*>(ak);
        uint4 a0 = a4[0], a1 = a4[1];
        const uint4* b4 = reinterpret_cast<const uint4*>(bk);
        uint4 b0 = b4[0], b1v = b4[1];
        uint4* ad = reinterpret_cast<uint4*>(adst[buf]); ad[0] = a0; ad[1] = a1;
        uint4* bd = reinterpret_cast<uint4*>(bdst[buf]); bd[0] = b0; bd[1] = b1v;
#endif
    };
    auto compute = [&](int buf) {
        Frag16 bq[2];
#pragma unroll
        for (int j = 0; j < 2; ++j) {
            const int col = wn + j * 16 + lrow, kb = half * 16;
            bq[j].u[0] = *reinterpret_cast<const uint4*>(&Bs[buf][col * 32 + kb]);
            bq[j].u[1] = *reinterpret_cast<const uint4*>(&Bs[buf][col * 32 + kb + 8]);
        }
#pragma unroll
        for (int i = 0; i < 4; ++i) {
            Frag16 a;   // lane halves hold K {0..15} / {16..31}
            const int rw = wm + i * 16 + lrow, ka = half * 16;
            a.u[0] = *reinterpret_cast<const uint4*>(&As[buf][rw * 32 + ka]);
            a.u[1] = *reinterpret_cast<const uint4*>(&As[buf][rw * 32 + ka + 8]);
#pragma unroll
            for (int j = 0; j < 2; ++j)
                acc[i][j] = __builtin_amdgcn_wmma_f32_16x16x32_bf16(
                    false, a.v, false, bq[j].v, (short)0, acc[i][j], false, false);
        }
    };

    constexpr int NK = Hh / 32;
    stage(0, 0);
    int p = 0;
    for (int kt = 0; kt < NK; ++kt) {
        wait_async_lds();               // my async tile writes landed in LDS
        __syncthreads();                // everyone's writes visible
        if (kt + 1 < NK) stage(kt + 1, p ^ 1);
        compute(p);
        p ^= 1;
    }

    // epilogue: relu(acc + b1) -> bf16 act rows
#pragma unroll
    for (int j = 0; j < 2; ++j) {
        const int gn = n0 + wn + j * 16 + lrow;
        const float bj = b1[(size_t)e * Ff + gn];
#pragma unroll
        for (int i = 0; i < 4; ++i) {
            const int rbase = wm + i * 16 + half * 8;
#pragma unroll
            for (int r = 0; r < 8; ++r) {
                const int rr = rbase + r;
                if (rr < mrows) {
                    float v = acc[i][j][r] + bj;
                    v = v > 0.f ? v : 0.f;
                    act[(size_t)(pbase + rr) * Ff + gn] = (bf16_t)v;
                }
            }
        }
    }
}

// ---------------------------------------------------------------------------
// 7) GEMM2: Y[p, h] = act[p] @ W2[e] + b2[e]   (fp32 out)
//    W2t pre-packed [E][H][F] (N-major).
// ---------------------------------------------------------------------------
__global__ __launch_bounds__(256) void gemm2_wmma_kernel(
    const bf16_t* __restrict__ act, const bf16_t* __restrict__ W2t,
    const float* __restrict__ b2, const int* __restrict__ counts,
    const int* __restrict__ offs, float* __restrict__ Y) {
    const int e    = blockIdx.x >> 7;
    const int mt   = blockIdx.x & 127;
    const int n0   = blockIdx.y * 128;
    const int cnt  = counts[e];
    const int row0 = mt * 128;
    if (row0 >= cnt) return;
    const int mrows = min(128, cnt - row0);
    const int pbase = offs[e] + row0;

    __shared__ bf16_t As[2][128 * 32];
    __shared__ bf16_t Bs[2][128 * 32];

    const int tid = threadIdx.x;
    const int wid = tid >> 5, lane = tid & 31;
    const int wm = (wid & 1) * 64, wn = (wid >> 1) * 32;
    const int lrow = lane & 15, half = lane >> 4;

    v8f acc[4][2];
#pragma unroll
    for (int i = 0; i < 4; ++i)
#pragma unroll
        for (int j = 0; j < 2; ++j)
#pragma unroll
            for (int r = 0; r < 8; ++r) acc[i][j][r] = 0.0f;

    const int sr = tid >> 1, sh = tid & 1;
    const int arow = pbase + ((sr < mrows) ? sr : 0);
    const bf16_t* aptr = act + (size_t)arow * Ff + sh * 16;
    const bf16_t* bptr = W2t + (size_t)e * Hh * Ff + (size_t)(n0 + sr) * Ff + sh * 16;
    bf16_t* adst[2] = { &As[0][sr * 32 + sh * 16], &As[1][sr * 32 + sh * 16] };
    bf16_t* bdst[2] = { &Bs[0][sr * 32 + sh * 16], &Bs[1][sr * 32 + sh * 16] };

    auto stage = [&](int kt, int buf) {
        const bf16_t* ak = aptr + kt * 32;
        const bf16_t* bk = bptr + kt * 32;
        __builtin_prefetch(bk + 64, 0, 0);
#if MOR_ASYNC_LDS
        async_b128(ak,     adst[buf]);
        async_b128(ak + 8, adst[buf] + 8);
        async_b128(bk,     bdst[buf]);
        async_b128(bk + 8, bdst[buf] + 8);
#else
        const uint4* a4 = reinterpret_cast<const uint4*>(ak);
        uint4 a0 = a4[0], a1 = a4[1];
        const uint4* b4 = reinterpret_cast<const uint4*>(bk);
        uint4 b0 = b4[0], b1v = b4[1];
        uint4* ad = reinterpret_cast<uint4*>(adst[buf]); ad[0] = a0; ad[1] = a1;
        uint4* bd = reinterpret_cast<uint4*>(bdst[buf]); bd[0] = b0; bd[1] = b1v;
#endif
    };
    auto compute = [&](int buf) {
        Frag16 bq[2];
#pragma unroll
        for (int j = 0; j < 2; ++j) {
            const int col = wn + j * 16 + lrow, kb = half * 16;
            bq[j].u[0] = *reinterpret_cast<const uint4*>(&Bs[buf][col * 32 + kb]);
            bq[j].u[1] = *reinterpret_cast<const uint4*>(&Bs[buf][col * 32 + kb + 8]);
        }
#pragma unroll
        for (int i = 0; i < 4; ++i) {
            Frag16 a;
            const int rw = wm + i * 16 + lrow, ka = half * 16;
            a.u[0] = *reinterpret_cast<const uint4*>(&As[buf][rw * 32 + ka]);
            a.u[1] = *reinterpret_cast<const uint4*>(&As[buf][rw * 32 + ka + 8]);
#pragma unroll
            for (int j = 0; j < 2; ++j)
                acc[i][j] = __builtin_amdgcn_wmma_f32_16x16x32_bf16(
                    false, a.v, false, bq[j].v, (short)0, acc[i][j], false, false);
        }
    };

    constexpr int NK = Ff / 32;
    stage(0, 0);
    int p = 0;
    for (int kt = 0; kt < NK; ++kt) {
        wait_async_lds();
        __syncthreads();
        if (kt + 1 < NK) stage(kt + 1, p ^ 1);
        compute(p);
        p ^= 1;
    }

#pragma unroll
    for (int j = 0; j < 2; ++j) {
        const int gn = n0 + wn + j * 16 + lrow;
        const float bj = b2[(size_t)e * Hh + gn];
#pragma unroll
        for (int i = 0; i < 4; ++i) {
            const int rbase = wm + i * 16 + half * 8;
#pragma unroll
            for (int r = 0; r < 8; ++r) {
                const int rr = rbase + r;
                if (rr < mrows)
                    Y[(size_t)(pbase + rr) * Hh + gn] = acc[i][j][r] + bj;
            }
        }
    }
}

// ---------------------------------------------------------------------------
// 8) combine: out = x + s0*Y0 + s1*Y1 - s0*c[i0] - s1*c[i1] + (s0+s1)*c_total
// ---------------------------------------------------------------------------
__global__ __launch_bounds__(256) void combine_kernel(
    const float* __restrict__ x, const float* __restrict__ Y,
    const float* __restrict__ scores, const int* __restrict__ topi,
    const int* __restrict__ pos, const float* __restrict__ c,
    const float* __restrict__ ct, float* __restrict__ out) {
    const int t = blockIdx.x, tid = threadIdx.x;
    const float s0 = scores[2 * t], s1 = scores[2 * t + 1];
    const int i0 = topi[2 * t], i1 = topi[2 * t + 1];
    const int p0 = pos[2 * t],  p1 = pos[2 * t + 1];
    const float* y0 = Y + (size_t)p0 * Hh;
    const float* y1 = Y + (size_t)p1 * Hh;
    const float* c0 = c + (size_t)i0 * Hh;
    const float* c1 = c + (size_t)i1 * Hh;
    const float* xr = x + (size_t)t * Hh;
    float* o = out + (size_t)t * Hh;
#pragma unroll
    for (int i = 0; i < 4; ++i) {
        const int h = tid + 256 * i;
        o[h] = xr[h] + s0 * y0[h] + s1 * y1[h]
             - s0 * c0[h] - s1 * c1[h] + (s0 + s1) * ct[h];
    }
}

// ---------------------------------------------------------------------------
extern "C" void kernel_launch(void* const* d_in, const int* in_sizes, int n_in,
                              void* d_out, int out_size, void* d_ws, size_t ws_size,
                              hipStream_t stream) {
    const float* x   = (const float*)d_in[0];
    const float* gmm = (const float*)d_in[1];
    const float* bta = (const float*)d_in[2];
    const float* Wr  = (const float*)d_in[3];
    const float* br  = (const float*)d_in[4];
    const float* W1  = (const float*)d_in[5];
    const float* b1  = (const float*)d_in[6];
    const float* W2  = (const float*)d_in[7];
    const float* b2  = (const float*)d_in[8];
    float* out = (float*)d_out;

    char* ws = (char*)d_ws;
    size_t cur = 0;
    auto carve = [&](size_t bytes) -> char* {
        char* p = ws + cur;
        cur = (cur + bytes + 255) & ~(size_t)255;
        return p;
    };
    bf16_t* xnb  = (bf16_t*)carve((size_t)T * Hh * sizeof(bf16_t));       //  32 MB
    bf16_t* W1t  = (bf16_t*)carve((size_t)Ee * Hh * Ff * sizeof(bf16_t)); //  64 MB
    bf16_t* W2t  = (bf16_t*)carve((size_t)Ee * Ff * Hh * sizeof(bf16_t)); //  64 MB
    bf16_t* actB = (bf16_t*)carve((size_t)TK * Ff * sizeof(bf16_t));      // 256 MB
    float*  Ybuf = (float*)carve((size_t)TK * Hh * sizeof(float));        // 128 MB
    float*  scor = (float*)carve((size_t)T * Kk * sizeof(float));
    int*    topi = (int*)carve((size_t)T * Kk * sizeof(int));
    int*    pos  = (int*)carve((size_t)T * Kk * sizeof(int));
    int*    tok  = (int*)carve((size_t)TK * sizeof(int));
    int*    cnts = (int*)carve(Ee * sizeof(int));
    int*    cnt2 = (int*)carve(Ee * sizeof(int));
    int*    offs = (int*)carve(Ee * sizeof(int));
    float*  cvec = (float*)carve((size_t)Ee * Hh * sizeof(float));
    float*  ctot = (float*)carve((size_t)Hh * sizeof(float));

    init_counts_kernel<<<1, 32, 0, stream>>>(cnts);
    // W1 [E][H][F] -> W1t [E][F][H];  W2 [E][F][H] -> W2t [E][H][F]
    transpose_pack_kernel<<<dim3(Ff / 32, Hh / 32, Ee), 256, 0, stream>>>(W1, W1t, Hh, Ff);
    transpose_pack_kernel<<<dim3(Hh / 32, Ff / 32, Ee), 256, 0, stream>>>(W2, W2t, Ff, Hh);
    ln_router_kernel<<<T, 256, 0, stream>>>(x, gmm, bta, Wr, br,
                                            xnb, scor, topi, cnts);
    offsets_kernel<<<1, 1, 0, stream>>>(cnts, offs, cnt2);
    build_list_kernel<<<T / 256, 256, 0, stream>>>(topi, offs, cnt2, tok, pos);
    c_kernel<<<(Ee * Hh) / 256, 256, 0, stream>>>(b1, W2, b2, cvec);
    ctot_kernel<<<Hh / 256, 256, 0, stream>>>(cvec, ctot);

    // GEMM1: grid.x = E * (T/128) = 1024, grid.y = F/128 = 32
    gemm1_wmma_kernel<<<dim3(Ee * (T / 128), Ff / 128), 256, 0, stream>>>(
        xnb, W1t, b1, tok, cnts, offs, actB);
    // GEMM2: grid.y = H/128 = 8
    gemm2_wmma_kernel<<<dim3(Ee * (T / 128), Hh / 128), 256, 0, stream>>>(
        actB, W2t, b2, cnts, offs, Ybuf);

    combine_kernel<<<T, 256, 0, stream>>>(x, Ybuf, scor, topi, pos,
                                          cvec, ctot, out);
}
